// SalienceAwarePlanner_84112639525065
// MI455X (gfx1250) — matrix operand order repaired and checked by reading, not compiled, not run on hardware
//
#include <hip/hip_runtime.h>
#include <math.h>

typedef __attribute__((ext_vector_type(16))) _Float16 v16h;
typedef __attribute__((ext_vector_type(8)))  float    v8f;
typedef __attribute__((ext_vector_type(4)))  float    f4;

namespace {
constexpr int kB = 16, kT = 4096, kH = 1024, kS = 16, kP = 6, kNH = 8, kDH = 128, kNL = 2;
constexpr float kEps = 1e-5f;
}

__device__ __forceinline__ float gelu_exact(float x) {
  return 0.5f * x * (1.0f + erff(x * 0.70710678118654752f));
}

// blockDim.x must be 256
__device__ __forceinline__ float block_reduce_sum(float v, float* red) {
  const int t = threadIdx.x;
  red[t] = v; __syncthreads();
  for (int s = 128; s > 0; s >>= 1) {
    if (t < s) red[t] += red[t + s];
    __syncthreads();
  }
  float r = red[0];
  __syncthreads();
  return r;
}

// ---------------------------------------------------------------------------
// C[M,N] = op( A[M,K] @ W[N,K]^T + bias[N] (+ resid[M,N]) )
// One wave computes a 16x64 strip: one M-tile x four N-tiles, four
// v_wmma_f32_16x16x32_f16 per K-step sharing one A load/convert.
// Software-pipelined: A preloaded one K-step ahead; B tile t+1's float4 loads
// are issued before tile t's f32->f16 conversion (done from copies) so loads
// overlap the cvt stream and the WMMA instead of wait-to-zero serialization.
// A lane layout (ISA 7.12.2, 16-bit A 16x32): lane m=lane&15, halves 0..7 ->
// K = kbA..kbA+7, halves 8..15 -> K = kbA+16..kbA+23, kbA = lane<16 ? 0 : 8.
// B lane layout (16-bit B 32x16): lane n=lane&15, halves 0..7 ->
// K = kbB..kbB+7, halves 8..15 -> K = kbB+16..kbB+23, kbB = lane<16 ? 0 : 8.
// D layout: VGPR v -> M = v + (lane<16 ? 0 : 8), N = lane&15.
// Requires M % 16 == 0, N % 64 == 0, K % 32 == 0, 16B-aligned A/W rows.
// ---------------------------------------------------------------------------
__global__ void wmma_gemm_xwT(const float* __restrict__ A, const float* __restrict__ W,
                              const float* __restrict__ bias, const float* __restrict__ resid,
                              float* __restrict__ C, int M, int N, int K, int op) {
  const int lane   = threadIdx.x & 31;
  const int wave   = (blockIdx.x * blockDim.x + threadIdx.x) >> 5;
  const int tilesN = N >> 6;                      // groups of 4 N-tiles
  const int tiles  = (M >> 4) * tilesN;
  if (wave >= tiles) return;                      // uniform per wave: EXEC stays all-ones
  const int tm  = wave / tilesN, tg = wave - tm * tilesN;
  const int l15 = lane & 15;
  const int kbA = (lane < 16) ? 0 : 8;
  const int kbB = (lane < 16) ? 0 : 8;
  const f4* ap = (const f4*)(A + (size_t)(tm * 16 + l15) * K + kbA);
  const f4* wk = (const f4*)(W + (size_t)(tg * 64 + l15) * K + kbB);
  const size_t tileStep = (size_t)4 * K;          // 16 W-rows = 16*K floats, in f4 units

  v8f acc[4];
#pragma unroll
  for (int t = 0; t < 4; ++t)
#pragma unroll
    for (int v = 0; v < 8; ++v) acc[t][v] = 0.f;

  // preload first A chunk and first B tile
  f4 a0 = ap[0], a1 = ap[1], a2 = ap[4], a3 = ap[5];
  f4 b0 = wk[0], b1 = wk[1], b2 = wk[4], b3 = wk[5];

  for (int k0 = 0; k0 < K; k0 += 32) {
    const bool last = (k0 + 32 >= K);
    // stage current A, issue next K-step's A loads
    f4 ca0 = a0, ca1 = a1, ca2 = a2, ca3 = a3;
    if (!last) { const f4* an = ap + 8; a0 = an[0]; a1 = an[1]; a2 = an[4]; a3 = an[5]; }
    v16h ah;
#pragma unroll
    for (int i = 0; i < 4; ++i) {
      ah[i]      = (_Float16)ca0[i];
      ah[i + 4]  = (_Float16)ca1[i];
      ah[i + 8]  = (_Float16)ca2[i];
      ah[i + 12] = (_Float16)ca3[i];
    }
#pragma unroll
    for (int t = 0; t < 4; ++t) {
      // stage current B tile, issue next B tile's loads before converting
      f4 c0 = b0, c1 = b1, c2 = b2, c3 = b3;
      if (!(last && t == 3)) {
        const f4* wn = (t == 3) ? (wk + 8) : (wk + (size_t)(t + 1) * tileStep);
        b0 = wn[0]; b1 = wn[1]; b2 = wn[4]; b3 = wn[5];
      }
      v16h bh;
#pragma unroll
      for (int j = 0; j < 4; ++j) {
        bh[j]      = (_Float16)c0[j];
        bh[j + 4]  = (_Float16)c1[j];
        bh[j + 8]  = (_Float16)c2[j];
        bh[j + 12] = (_Float16)c3[j];
      }
      acc[t] = __builtin_amdgcn_wmma_f32_16x16x32_f16(false, ah, false, bh, (short)0, acc[t],
                                                      false, false);
    }
    ap += 8; wk += 8;
  }

  const int rbase = tm * 16 + ((lane < 16) ? 0 : 8);
#pragma unroll
  for (int t = 0; t < 4; ++t) {
    const int col = tg * 64 + t * 16 + l15;
    const float bv = bias ? bias[col] : 0.f;
#pragma unroll
    for (int v = 0; v < 8; ++v) {
      const int r = rbase + v;
      float x = acc[t][v] + bv;
      if (resid) x += resid[(size_t)r * N + col];
      if (op == 1) x = gelu_exact(x);
      C[(size_t)r * N + col] = x;
    }
  }
}

// Generic tiny-sequence MHA core: per-(batch,head) scores, mask, softmax, A@V.
// Q rows: b*Mq + i (head slice qoff + h*128). K/V rows: b*Mk + j (koff/voff).
__global__ void mha_core(const float* __restrict__ Q, int ldq, int qoff,
                         const float* __restrict__ KV, int ldkv, int koff, int voff,
                         const float* __restrict__ cnt,
                         float* __restrict__ O, int ldo, int Mq, int Mk) {
  const int b = blockIdx.x / kNH;
  const int h = blockIdx.x % kNH;
  const int tid = threadIdx.x;
  __shared__ float sc[16 * 16];
  const float scale = 0.0883883476483184406f;   // 1/sqrt(128)
  for (int idx = tid; idx < Mq * Mk; idx += blockDim.x) {
    const int i = idx / Mk, j = idx - i * Mk;
    const float* qp = Q  + (size_t)(b * Mq + i) * ldq  + qoff + h * kDH;
    const float* kp = KV + (size_t)(b * Mk + j) * ldkv + koff + h * kDH;
    float s = 0.f;
    for (int d = 0; d < kDH; ++d) s += qp[d] * kp[d];
    s *= scale;
    if (cnt && cnt[b * kS + j] <= 0.f) s = -1e30f;
    sc[i * 16 + j] = s;
  }
  __syncthreads();
  if (tid < Mq) {
    float mx = -3.4e38f;
    for (int j = 0; j < Mk; ++j) mx = fmaxf(mx, sc[tid * 16 + j]);
    float sum = 0.f;
    for (int j = 0; j < Mk; ++j) { float e = expf(sc[tid * 16 + j] - mx); sc[tid * 16 + j] = e; sum += e; }
    const float inv = 1.f / sum;
    for (int j = 0; j < Mk; ++j) sc[tid * 16 + j] *= inv;
  }
  __syncthreads();
  for (int idx = tid; idx < Mq * kDH; idx += blockDim.x) {
    const int i = idx >> 7, d = idx & 127;
    float s = 0.f;
    for (int j = 0; j < Mk; ++j)
      s += sc[i * 16 + j] * KV[(size_t)(b * Mk + j) * ldkv + voff + h * kDH + d];
    O[(size_t)(b * Mq + i) * ldo + h * kDH + d] = s;
  }
}

__global__ void seg_bounds_kernel(const int* __restrict__ mask, int* __restrict__ bounds) {
  const int b = blockIdx.x;
  __shared__ int redi[256];
  int s = 0;
  for (int t = threadIdx.x; t < kT; t += 256) s += mask[b * kT + t];
  redi[threadIdx.x] = s; __syncthreads();
  for (int st = 128; st > 0; st >>= 1) {
    if (threadIdx.x < st) redi[threadIdx.x] += redi[threadIdx.x + st];
    __syncthreads();
  }
  const int L = redi[0];                          // s*L <= 65536 < 2^24: exact match to jnp.floor
  if (threadIdx.x <= kS) bounds[b * (kS + 1) + threadIdx.x] = (threadIdx.x * L) / kS;
}

// HBM-bound phase: grid (B*S, 4); each block streams one segment's contiguous rows.
__global__ void seg_pool_kernel(const float* __restrict__ tok, const int* __restrict__ bounds,
                                float* __restrict__ seg, float* __restrict__ cnt) {
  const int bs = blockIdx.x;
  const int b = bs >> 4, s = bs & 15;
  const int t0 = bounds[b * (kS + 1) + s], t1 = bounds[b * (kS + 1) + s + 1];
  const int ch = blockIdx.y * 256 + threadIdx.x;
  const float* base = tok + ((size_t)b * kT) * kH + ch;
  float acc = 0.f;
  for (int t = t0; t < t1; ++t) {
    acc += base[(size_t)t * kH];
    if (t + 8 < t1) __builtin_prefetch(base + (size_t)(t + 8) * kH, 0, 1);
  }
  const float c = (float)(t1 - t0);
  seg[(size_t)bs * kH + ch] = acc / fmaxf(c, 1.f);
  if (blockIdx.y == 0 && threadIdx.x == 0) cnt[bs] = c;
}

__global__ void ln_kernel(const float* __restrict__ x, const float* __restrict__ g,
                          const float* __restrict__ bta, float* __restrict__ y) {
  __shared__ float red[256];
  const int row = blockIdx.x;
  const float* xr = x + (size_t)row * kH;
  float s = 0.f;
  for (int i = threadIdx.x; i < kH; i += 256) s += xr[i];
  const float m = block_reduce_sum(s, red) * (1.f / kH);
  s = 0.f;
  for (int i = threadIdx.x; i < kH; i += 256) { const float d = xr[i] - m; s += d * d; }
  const float var = block_reduce_sum(s, red) * (1.f / kH);
  const float inv = rsqrtf(var + kEps);
  float* yr = y + (size_t)row * kH;
  for (int i = threadIdx.x; i < kH; i += 256) yr[i] = (xr[i] - m) * inv * g[i] + bta[i];
}

__global__ void rowdot_kernel(const float* __restrict__ A, const float* __restrict__ w,
                              const float* __restrict__ bias, float* __restrict__ out) {
  __shared__ float red[256];
  const int row = blockIdx.x;
  float s = 0.f;
  for (int i = threadIdx.x; i < kH; i += 256) s += A[(size_t)row * kH + i] * w[i];
  const float tot = block_reduce_sum(s, red);
  if (threadIdx.x == 0) out[row] = tot + bias[0];
}

__global__ void sal_softmax_kernel(const float* __restrict__ logits, const float* __restrict__ cnt,
                                   float* __restrict__ sal, float* __restrict__ ent_part) {
  const int b = blockIdx.x;
  if (threadIdx.x == 0) {
    float v[kS];
    float mx = -3.4e38f;
    for (int s = 0; s < kS; ++s) {
      float l = logits[b * kS + s];
      if (cnt[b * kS + s] <= 0.f) l = -10000.f;
      v[s] = l; mx = fmaxf(mx, l);
    }
    float sum = 0.f;
    for (int s = 0; s < kS; ++s) { v[s] = expf(v[s] - mx); sum += v[s]; }
    float ent = 0.f;
    for (int s = 0; s < kS; ++s) {
      const float p = v[s] / sum;
      sal[b * kS + s] = p;
      ent -= p * logf(p + 1e-8f);
    }
    ent_part[b] = ent;
  }
}

__global__ void scale_rows_kernel(const float* __restrict__ x, const float* __restrict__ s,
                                  float* __restrict__ y, int total) {
  for (int i = blockIdx.x * blockDim.x + threadIdx.x; i < total; i += gridDim.x * blockDim.x)
    y[i] = x[i] * s[i >> 10];                     // kH == 1024
}

__global__ void bcast_q_kernel(const float* __restrict__ pq, float* __restrict__ qin) {
  const int bp = blockIdx.x;                      // 0..B*P-1
  const int p = bp % kP;
  for (int i = threadIdx.x; i < kH; i += 256)
    qin[(size_t)bp * kH + i] = pq[p * kH + i];
}

__global__ void redundancy_kernel(const float* __restrict__ plan, float* __restrict__ red_part) {
  __shared__ float red[256];
  __shared__ float nrm[kP];
  __shared__ float pairv;
  const int b = blockIdx.x;
  const float* Pb = plan + (size_t)b * kP * kH;
  for (int p = 0; p < kP; ++p) {
    float s = 0.f;
    for (int i = threadIdx.x; i < kH; i += 256) { const float v = Pb[p * kH + i]; s += v * v; }
    const float tot = block_reduce_sum(s, red);
    if (threadIdx.x == 0) nrm[p] = fmaxf(sqrtf(tot), 1e-12f);
  }
  __syncthreads();
  float total = 0.f;
  for (int p = 0; p < kP; ++p)
    for (int q = p + 1; q < kP; ++q) {
      float s = 0.f;
      for (int i = threadIdx.x; i < kH; i += 256) s += Pb[p * kH + i] * Pb[q * kH + i];
      const float tot = block_reduce_sum(s, red);
      if (threadIdx.x == 0) { const float sim = tot / (nrm[p] * nrm[q]); pairv = 2.f * sim * sim; }
      __syncthreads();
      total += pairv;                             // both (p,q) and (q,p) of off-diag
      __syncthreads();
    }
  if (threadIdx.x == 0) red_part[b] = total;
}

__global__ void finalize_kernel(const float* __restrict__ ent_part, const float* __restrict__ red_part,
                                float* __restrict__ out2) {
  if (blockIdx.x == 0 && threadIdx.x == 0) {
    float e = 0.f, r = 0.f;
    for (int b = 0; b < kB; ++b) { e += ent_part[b]; r += red_part[b]; }
    out2[0] = e / (float)kB;
    out2[1] = r / (float)(kB * kP * (kP - 1));
  }
}

extern "C" void kernel_launch(void* const* d_in, const int* in_sizes, int n_in,
                              void* d_out, int out_size, void* d_ws, size_t ws_size,
                              hipStream_t stream) {
  (void)in_sizes; (void)n_in; (void)out_size; (void)ws_size;
  const float* tok      = (const float*)d_in[0];
  const int*   mask     = (const int*)  d_in[1];
  const float* sa_in_w  = (const float*)d_in[2];
  const float* sa_in_b  = (const float*)d_in[3];
  const float* sa_out_w = (const float*)d_in[4];
  const float* sa_out_b = (const float*)d_in[5];
  const float* ln_g     = (const float*)d_in[6];
  const float* ln_b     = (const float*)d_in[7];
  const float* sal_w1   = (const float*)d_in[8];
  const float* sal_b1   = (const float*)d_in[9];
  const float* sal_w2   = (const float*)d_in[10];
  const float* sal_b2   = (const float*)d_in[11];
  const float* plan_q   = (const float*)d_in[12];
  const float* qa_in_w  = (const float*)d_in[13];
  const float* qa_in_b  = (const float*)d_in[14];
  const float* qa_out_w = (const float*)d_in[15];
  const float* qa_out_b = (const float*)d_in[16];
  const float* r_in_w   = (const float*)d_in[17];
  const float* r_in_b   = (const float*)d_in[18];
  const float* r_out_w  = (const float*)d_in[19];
  const float* r_out_b  = (const float*)d_in[20];
  const float* r_ln1_g  = (const float*)d_in[21];
  const float* r_ln1_b  = (const float*)d_in[22];
  const float* r_ln2_g  = (const float*)d_in[23];
  const float* r_ln2_b  = (const float*)d_in[24];
  const float* r_w1     = (const float*)d_in[25];
  const float* r_b1     = (const float*)d_in[26];
  const float* r_w2     = (const float*)d_in[27];
  const float* r_b2     = (const float*)d_in[28];

  float* out  = (float*)d_out;
  float* plan = out;                               // [B,P,H] (written before first read)
  float* sal  = out + (size_t)kB * kP * kH;        // [B,S]
  float* scal = sal + (size_t)kB * kS;             // [entropy, redundancy]

  int*   bounds = (int*)d_ws;                      // B*(S+1) ints
  float* f = (float*)((char*)d_ws + 4096);
  size_t off = 0;
  auto alloc = [&](size_t n) { float* p = f + off; off += (n + 63) & ~(size_t)63; return p; };
  float* cnt     = alloc((size_t)kB * kS);
  float* segp    = alloc((size_t)kB * kS * kH);
  float* qkv_seg = alloc((size_t)kB * kS * 3 * kH);
  float* attnO   = alloc((size_t)kB * kS * kH);
  float* seg_sum = alloc((size_t)kB * kS * kH);
  float* seg     = alloc((size_t)kB * kS * kH);
  float* h_sal   = alloc((size_t)kB * kS * kH);
  float* logits  = alloc((size_t)kB * kS);
  float* ent_p   = alloc(kB);
  float* red_p   = alloc(kB);
  float* wsegb   = alloc((size_t)kB * kS * kH);
  float* qin     = alloc((size_t)kB * kP * kH);
  float* qbuf    = alloc((size_t)kB * kP * kH);
  float* kvbuf   = alloc((size_t)kB * kS * 2 * kH);
  float* attnP   = alloc((size_t)kB * kP * kH);
  float* ybuf    = alloc((size_t)kB * kP * kH);
  float* qkv_r   = alloc((size_t)kB * kP * 3 * kH);
  float* attnR   = alloc((size_t)kB * kP * kH);
  float* h1      = alloc((size_t)kB * kP * 4 * kH);

  auto gemm = [&](const float* A, const float* W, const float* bias, const float* resid,
                  float* C, int M, int N, int K, int op) {
    const int tiles  = (M / 16) * (N / 64);        // 16x64 strip per wave
    const int blocks = (tiles + 7) / 8;            // 8 waves / block (256 thr)
    wmma_gemm_xwT<<<blocks, 256, 0, stream>>>(A, W, bias, resid, C, M, N, K, op);
  };

  // 1) segment pooling (dominant HBM traffic: token_states read exactly once)
  seg_bounds_kernel<<<kB, 256, 0, stream>>>(mask, bounds);
  seg_pool_kernel<<<dim3(kB * kS, kH / 256), 256, 0, stream>>>(tok, bounds, segp, cnt);
  // 2) segment self-attention
  gemm(segp, sa_in_w, sa_in_b, nullptr, qkv_seg, kB * kS, 3 * kH, kH, 0);
  mha_core<<<kB * kNH, 256, 0, stream>>>(qkv_seg, 3 * kH, 0, qkv_seg, 3 * kH, kH, 2 * kH,
                                         cnt, attnO, kH, kS, kS);
  gemm(attnO, sa_out_w, sa_out_b, segp, seg_sum, kB * kS, kH, kH, 0);
  ln_kernel<<<kB * kS, 256, 0, stream>>>(seg_sum, ln_g, ln_b, seg);
  // 3) salience
  gemm(seg, sal_w1, sal_b1, nullptr, h_sal, kB * kS, kH, kH, 1);
  rowdot_kernel<<<kB * kS, 256, 0, stream>>>(h_sal, sal_w2, sal_b2, logits);
  sal_softmax_kernel<<<kB, 32, 0, stream>>>(logits, cnt, sal, ent_p);
  scale_rows_kernel<<<256, 256, 0, stream>>>(seg, sal, wsegb, kB * kS * kH);
  // 4) plan-query cross attention
  bcast_q_kernel<<<kB * kP, 256, 0, stream>>>(plan_q, qin);
  gemm(qin, qa_in_w, qa_in_b, nullptr, qbuf, kB * kP, kH, kH, 0);
  gemm(wsegb, qa_in_w + (size_t)kH * kH, qa_in_b + kH, nullptr, kvbuf, kB * kS, 2 * kH, kH, 0);
  mha_core<<<kB * kNH, 256, 0, stream>>>(qbuf, kH, 0, kvbuf, 2 * kH, 0, kH,
                                         cnt, attnP, kH, kP, kS);
  gemm(attnP, qa_out_w, qa_out_b, nullptr, plan, kB * kP, kH, kH, 0);
  // 5) norm-first transformer encoder layers
  for (int l = 0; l < kNL; ++l) {
    ln_kernel<<<kB * kP, 256, 0, stream>>>(plan, r_ln1_g + l * kH, r_ln1_b + l * kH, ybuf);
    gemm(ybuf, r_in_w + (size_t)l * 3 * kH * kH, r_in_b + (size_t)l * 3 * kH, nullptr,
         qkv_r, kB * kP, 3 * kH, kH, 0);
    mha_core<<<kB * kNH, 256, 0, stream>>>(qkv_r, 3 * kH, 0, qkv_r, 3 * kH, kH, 2 * kH,
                                           nullptr, attnR, kH, kP, kP);
    gemm(attnR, r_out_w + (size_t)l * kH * kH, r_out_b + (size_t)l * kH, plan,
         plan, kB * kP, kH, kH, 0);
    ln_kernel<<<kB * kP, 256, 0, stream>>>(plan, r_ln2_g + l * kH, r_ln2_b + l * kH, ybuf);
    gemm(ybuf, r_w1 + (size_t)l * 4 * kH * kH, r_b1 + (size_t)l * 4 * kH, nullptr,
         h1, kB * kP, 4 * kH, kH, 1);
    gemm(h1, r_w2 + (size_t)l * kH * 4 * kH, r_b2 + (size_t)l * kH, plan,
         plan, kB * kP, kH, 4 * kH, 0);
  }
  // 6) scalars
  redundancy_kernel<<<kB, 256, 0, stream>>>(plan, red_p);
  finalize_kernel<<<1, 32, 0, stream>>>(ent_p, red_p, scal);
}
